// BoostedCausalAttention_223338299735
// MI455X (gfx1250) — matrix operand order repaired and checked by reading, not compiled, not run on hardware
//
#include <hip/hip_runtime.h>
#include <hip/hip_bf16.h>

// ---------------------------------------------------------------------------
// BoostedCausalAttention for MI455X (gfx1250, wave32, WMMA bf16 + async LDS)
// ---------------------------------------------------------------------------

typedef __bf16 bf16;
typedef __attribute__((ext_vector_type(16))) __bf16 v16bf;
typedef __attribute__((ext_vector_type(8)))  __bf16 bf16x8;
typedef __attribute__((ext_vector_type(8)))  float  v8f;
typedef __attribute__((ext_vector_type(4)))  int    v4i;

#define BB  2
#define TT  2048
#define DD  1024
#define HH  16
#define DHH 64
#define MM  (BB * TT)            // 4096 rows total
static constexpr float ATT_SCALE = 0.125f;   // 1/sqrt(64)

// ---------------------------------------------------------------------------
// CDNA5 async global->LDS copy (ASYNCcnt path), with portable fallback
// ---------------------------------------------------------------------------
#if defined(__has_builtin)
#if __has_builtin(__builtin_amdgcn_global_load_async_to_lds_b128) && \
    __has_builtin(__builtin_amdgcn_s_wait_asynccnt)
#define HAVE_ASYNC_LDS 1
#endif
#endif
#ifndef HAVE_ASYNC_LDS
#define HAVE_ASYNC_LDS 0
#endif

typedef __attribute__((address_space(1))) v4i g_v4i;   // global int4
typedef __attribute__((address_space(3))) v4i l_v4i;   // LDS int4

__device__ __forceinline__ void async_copy16(const bf16* g, bf16* l) {
#if HAVE_ASYNC_LDS
  __builtin_amdgcn_global_load_async_to_lds_b128((g_v4i*)g, (l_v4i*)l, 0, 0);
#else
  *(bf16x8*)l = *(const bf16x8*)g;
#endif
}

__device__ __forceinline__ void wait_async() {
#if HAVE_ASYNC_LDS
  __builtin_amdgcn_s_wait_asynccnt(0);
#endif
}

// ---------------------------------------------------------------------------
// WMMA fragment helpers (layouts per CDNA5 ISA 7.12.2, wave32)
// ---------------------------------------------------------------------------
__device__ __forceinline__ v8f v8f_zero() {
  v8f z;
#pragma unroll
  for (int i = 0; i < 8; ++i) z[i] = 0.0f;
  return z;
}

// A-matrix 16x32 bf16: lane = m (mod 16), half = lane/16.
// elements 0..7 -> k = k0 + half*8 + i ; elements 8..15 -> k = k0 + 16 + half*8 + i
__device__ __forceinline__ v16bf frag_a_lds(const bf16* base, int ld, int m0, int k0) {
  const int lane = threadIdx.x & 31;
  const int row  = m0 + (lane & 15);
  const int half = lane >> 4;
  const bf16* p = base + row * ld + k0 + half * 8;
  union { v16bf v; bf16x8 h[2]; } u;
  u.h[0] = *(const bf16x8*)(p);
  u.h[1] = *(const bf16x8*)(p + 16);
  return u.v;
}

// B-matrix 32x16 bf16 fed from a row-major [n][k] tile: lane = n (mod 16),
// half = lane/16: elements 0..7 -> k = k0 + half*8 + i ; elements 8..15 -> k = k0 + 16 + half*8 + i
__device__ __forceinline__ v16bf frag_b_lds(const bf16* base, int ld, int n0, int k0) {
  const int lane = threadIdx.x & 31;
  const int row  = n0 + (lane & 15);
  const int half = lane >> 4;
  const bf16* p = base + row * ld + k0 + half * 8;
  union { v16bf v; bf16x8 h[2]; } u;
  u.h[0] = *(const bf16x8*)(p);
  u.h[1] = *(const bf16x8*)(p + 16);
  return u.v;
}

__device__ __forceinline__ v8f wmma_bf16(v16bf a, v16bf b, v8f c) {
  return __builtin_amdgcn_wmma_f32_16x16x32_bf16(
      /*neg_a=*/false, a, /*neg_b=*/false, b,
      /*c_mod=*/(short)0, c, /*reuse_a=*/false, /*reuse_b=*/false);
}

// ---------------------------------------------------------------------------
// Generic bf16 GEMM:  out = epilogue(A[M,K] @ Bt[N,K]^T + bias)
// 256 threads = 8 waves; block tile 128x128; wave tile 32x64; K step 32.
// Double-buffered LDS staged with async global->LDS copies (one barrier/step).
// EPI: 0 = store bf16, 1 = gated combine (pred + sigmoid(acc)*corr, bf16),
//      2 = store fp32
// ---------------------------------------------------------------------------
template <int EPI>
__global__ __launch_bounds__(256) void gemm_bf16_k(
    const bf16* __restrict__ A, const bf16* __restrict__ Bt,
    const float* __restrict__ bias, void* __restrict__ outp,
    const bf16* __restrict__ pred, const bf16* __restrict__ corr,
    int M, int N, int K) {
  __shared__ __align__(16) bf16 As[2][128][40];   // 128 rows x 32 k (pad 40)
  __shared__ __align__(16) bf16 Bs[2][128][40];   // 128 n-rows x 32 k

  const int tid    = threadIdx.x;
  const int wave   = tid >> 5;
  const int lane   = tid & 31;
  const int wave_m = wave >> 1;      // 0..3 -> 32 rows each
  const int wave_n = wave & 1;       // 0..1 -> 64 cols each
  const int m0 = blockIdx.x * 128;
  const int n0 = blockIdx.y * 128;

  v8f acc[2][4];
#pragma unroll
  for (int mi = 0; mi < 2; ++mi)
#pragma unroll
    for (int j = 0; j < 4; ++j) acc[mi][j] = v8f_zero();

  // stage tile kt into buffer buf: 128x32 bf16 per matrix = 512 x 16B chunks
  auto stage = [&](int k0, int buf) {
#pragma unroll
    for (int c = tid; c < 512; c += 256) {
      const int row = c >> 2;
      const int col = (c & 3) * 8;
      async_copy16(&A[(size_t)(m0 + row) * K + k0 + col], &As[buf][row][col]);
      async_copy16(&Bt[(size_t)(n0 + row) * K + k0 + col], &Bs[buf][row][col]);
    }
  };

  const int nk = K / 32;
  stage(0, 0);
  int buf = 0;
  for (int kt = 0; kt < nk; ++kt) {
    wait_async();
    __syncthreads();                       // tile kt resident; buf^1 free
    if (kt + 1 < nk) stage((kt + 1) * 32, buf ^ 1);

    const bf16* Ab = &As[buf][0][0];
    const bf16* Bb = &Bs[buf][0][0];
    v16bf afr[2];
#pragma unroll
    for (int mi = 0; mi < 2; ++mi)
      afr[mi] = frag_a_lds(Ab, 40, wave_m * 32 + mi * 16, 0);
#pragma unroll
    for (int j = 0; j < 4; ++j) {
      const v16bf bfr = frag_b_lds(Bb, 40, wave_n * 64 + j * 16, 0);
#pragma unroll
      for (int mi = 0; mi < 2; ++mi)
        acc[mi][j] = wmma_bf16(afr[mi], bfr, acc[mi][j]);
    }
    buf ^= 1;
  }

  // Epilogue: C layout -> m = +r + 8*(lane>=16), n = (lane&15)
  const int mb = m0 + wave_m * 32 + ((lane >> 4) << 3);
  const int nb = n0 + wave_n * 64 + (lane & 15);
#pragma unroll
  for (int mi = 0; mi < 2; ++mi) {
#pragma unroll
    for (int j = 0; j < 4; ++j) {
      const int n_g = nb + j * 16;
      const float bv = bias[n_g];
#pragma unroll
      for (int r = 0; r < 8; ++r) {
        const int m_g = mb + mi * 16 + r;
        const float v = acc[mi][j][r] + bv;
        const size_t o = (size_t)m_g * N + n_g;
        if (EPI == 0) {
          ((bf16*)outp)[o] = (bf16)v;
        } else if (EPI == 1) {
          const float g = 1.0f / (1.0f + __expf(-v));
          const float res = (float)pred[o] + g * (float)corr[o];
          ((bf16*)outp)[o] = (bf16)res;
        } else {
          ((float*)outp)[o] = v;
        }
      }
    }
  }
}

// ---------------------------------------------------------------------------
// Flash attention (causal), bf16 WMMA, fp32 online softmax.
// grid: (T/64, B*H); block 128 = 4 waves; wave owns 16 q-rows of the 64-row
// q tile. K/V processed in 64-key tiles up to the diagonal.
// qkv layout: [(b*T + t) * 3D + which*D + h*DH + d]
// ---------------------------------------------------------------------------
__global__ __launch_bounds__(128) void flash_attn_k(
    const bf16* __restrict__ qkv, bf16* __restrict__ outp) {
  __shared__ __align__(16) bf16 Qs[64][72];
  __shared__ __align__(16) bf16 Ks[64][72];
  __shared__ __align__(16) bf16 Vt[64][72];        // transposed: Vt[d][key]
  __shared__ __align__(16) bf16 Ps[4][16][72];     // per-wave P staging

  const int tid  = threadIdx.x;
  const int wave = tid >> 5;
  const int lane = tid & 31;
  const int qt = blockIdx.x;
  const int bh = blockIdx.y;
  const int b  = bh >> 4;
  const int h  = bh & 15;

  const size_t rs   = 3 * DD;
  const size_t base = ((size_t)b * TT) * rs + (size_t)h * DHH;

  // Load 64x64 Q tile (async path)
  for (int c = tid; c < 512; c += 128) {
    const int row = c >> 3;
    const int col = (c & 7) * 8;
    async_copy16(&qkv[base + (size_t)(qt * 64 + row) * rs + col], &Qs[row][col]);
  }

  v8f acc_o[4];
#pragma unroll
  for (int j = 0; j < 4; ++j) acc_o[j] = v8f_zero();
  float m_run[8], l_run[8];
#pragma unroll
  for (int r = 0; r < 8; ++r) { m_run[r] = -1e30f; l_run[r] = 0.0f; }

  const int nloc = lane & 15;
  const int mloc = wave * 16 + ((lane >> 4) << 3);

  for (int kt = 0; kt <= qt; ++kt) {
    // Stage K (async, row-major) and V (register transpose) tiles
    for (int c = tid; c < 512; c += 128) {
      const int row = c >> 3;
      const int col = (c & 7) * 8;
      const size_t g = base + (size_t)(kt * 64 + row) * rs + col;
      async_copy16(&qkv[g + DD], &Ks[row][col]);
      const bf16x8 vv = *(const bf16x8*)&qkv[g + 2 * DD];
#pragma unroll
      for (int i = 0; i < 8; ++i) Vt[col + i][row] = vv[i];
    }
    wait_async();
    __syncthreads();

    // S = Q @ K^T  (wave: 16 q-rows x 64 keys, DH=64 -> two K=32 steps)
    v8f s[4];
#pragma unroll
    for (int j = 0; j < 4; ++j) s[j] = v8f_zero();
#pragma unroll
    for (int ks = 0; ks < 2; ++ks) {
      const v16bf af = frag_a_lds(&Qs[0][0], 72, wave * 16, ks * 32);
#pragma unroll
      for (int j = 0; j < 4; ++j) {
        const v16bf bfr = frag_b_lds(&Ks[0][0], 72, j * 16, ks * 32);
        s[j] = wmma_bf16(af, bfr, s[j]);
      }
    }

    // Scale + causal mask (only diagonal tile needs element masking)
#pragma unroll
    for (int j = 0; j < 4; ++j) {
#pragma unroll
      for (int r = 0; r < 8; ++r) {
        float v = s[j][r] * ATT_SCALE;
        if (kt == qt && (j * 16 + nloc) > (mloc + r)) v = -1e30f;
        s[j][r] = v;
      }
    }

    // Online softmax: each 16x16 row lives in one VGPR across a 16-lane group
#pragma unroll
    for (int r = 0; r < 8; ++r) {
      float mx = fmaxf(fmaxf(s[0][r], s[1][r]), fmaxf(s[2][r], s[3][r]));
#pragma unroll
      for (int msk = 1; msk < 16; msk <<= 1)
        mx = fmaxf(mx, __shfl_xor(mx, msk, 32));
      const float mnew  = fmaxf(m_run[r], mx);
      const float alpha = __expf(m_run[r] - mnew);
      m_run[r] = mnew;
      float sum = 0.0f;
#pragma unroll
      for (int j = 0; j < 4; ++j) {
        const float p = __expf(s[j][r] - mnew);
        s[j][r] = p;
        sum += p;
      }
#pragma unroll
      for (int msk = 1; msk < 16; msk <<= 1) sum += __shfl_xor(sum, msk, 32);
      l_run[r] = l_run[r] * alpha + sum;
#pragma unroll
      for (int j = 0; j < 4; ++j) acc_o[j][r] *= alpha;
    }

    // Re-lay-out P (C layout) into A-fragment layout via per-wave LDS.
    // Same-wave DS ops are in-order: no barrier needed before reading back.
    const int prow = (lane >> 4) << 3;
#pragma unroll
    for (int j = 0; j < 4; ++j)
#pragma unroll
      for (int r = 0; r < 8; ++r)
        Ps[wave][prow + r][j * 16 + nloc] = (bf16)s[j][r];

    // O += P @ V   (keys: two K=32 steps; dh: four 16-col tiles)
#pragma unroll
    for (int ks = 0; ks < 2; ++ks) {
      const v16bf af = frag_a_lds(&Ps[wave][0][0], 72, 0, ks * 32);
#pragma unroll
      for (int j = 0; j < 4; ++j) {
        const v16bf bfr = frag_b_lds(&Vt[0][0], 72, j * 16, ks * 32);
        acc_o[j] = wmma_bf16(af, bfr, acc_o[j]);
      }
    }
    __syncthreads();
  }

  // Normalize and write out (row-major [M, D], head column offset h*DH)
  const size_t orow0 = (size_t)(b * TT + qt * 64 + wave * 16 + ((lane >> 4) << 3));
#pragma unroll
  for (int j = 0; j < 4; ++j)
#pragma unroll
    for (int r = 0; r < 8; ++r) {
      const float v = acc_o[j][r] / l_run[r];
      outp[(orow0 + r) * DD + h * DHH + j * 16 + (lane & 15)] = (bf16)v;
    }
}

// ---------------------------------------------------------------------------
// Elementwise helpers
// ---------------------------------------------------------------------------
__global__ void cvt_f32_bf16_k(const float* __restrict__ in,
                               bf16* __restrict__ out, int n) {
  const int i = blockIdx.x * 256 + threadIdx.x;
  if (i < n) out[i] = (bf16)in[i];
}

// Wt[n*K + k] = (bf16) W[k*N + n]
__global__ void transpose_w_k(const float* __restrict__ W,
                              bf16* __restrict__ Wt, int K, int N) {
  const int i = blockIdx.x * 256 + threadIdx.x;
  if (i < K * N) {
    const int n = i / K;
    const int k = i - n * K;
    Wt[i] = (bf16)W[(size_t)k * N + n];
  }
}

__global__ void resid_k(const float* __restrict__ x,
                        const bf16* __restrict__ pred,
                        bf16* __restrict__ out, int n) {
  const int i = blockIdx.x * 256 + threadIdx.x;
  if (i < n) out[i] = (bf16)(x[i] - (float)pred[i]);
}

__global__ void concat_k(const bf16* __restrict__ a, const bf16* __restrict__ b,
                         bf16* __restrict__ out, int n) {  // n = M*2D
  const int i = blockIdx.x * 256 + threadIdx.x;
  if (i < n) {
    const int m = i / (2 * DD);
    const int c = i - m * (2 * DD);
    out[i] = (c < DD) ? a[(size_t)m * DD + c] : b[(size_t)m * DD + (c - DD)];
  }
}

// ---------------------------------------------------------------------------
// Host-side launch
// ---------------------------------------------------------------------------
extern "C" void kernel_launch(void* const* d_in, const int* in_sizes, int n_in,
                              void* d_out, int out_size, void* d_ws,
                              size_t ws_size, hipStream_t stream) {
  (void)in_sizes; (void)n_in; (void)out_size; (void)ws_size;
  const float* x     = (const float*)d_in[0];
  const float* Wqkv0 = (const float*)d_in[1];
  const float* bqkv0 = (const float*)d_in[2];
  const float* Wqkv1 = (const float*)d_in[3];
  const float* bqkv1 = (const float*)d_in[4];
  const float* Wg    = (const float*)d_in[5];
  const float* bg    = (const float*)d_in[6];
  const float* Wo    = (const float*)d_in[7];
  const float* bo    = (const float*)d_in[8];

  char* ws = (char*)d_ws;
  size_t off = 0;
  auto alloc = [&](size_t bytes) {
    char* p = ws + off;
    off += (bytes + 255) & ~(size_t)255;
    return p;
  };
  bf16* xb     = (bf16*)alloc((size_t)MM * DD * 2);
  bf16* w0t    = (bf16*)alloc((size_t)3 * DD * DD * 2);
  bf16* w1t    = (bf16*)alloc((size_t)3 * DD * DD * 2);
  bf16* wgt    = (bf16*)alloc((size_t)DD * 2 * DD * 2);
  bf16* wot    = (bf16*)alloc((size_t)DD * DD * 2);
  bf16* qkvb   = (bf16*)alloc((size_t)MM * 3 * DD * 2);
  bf16* predb  = (bf16*)alloc((size_t)MM * DD * 2);
  bf16* corrb  = (bf16*)alloc((size_t)MM * DD * 2);
  bf16* residb = (bf16*)alloc((size_t)MM * DD * 2);
  bf16* catb   = (bf16*)alloc((size_t)MM * 2 * DD * 2);
  bf16* outb   = (bf16*)alloc((size_t)MM * DD * 2);

  const int TPB = 256;
  cvt_f32_bf16_k<<<(MM * DD + TPB - 1) / TPB, TPB, 0, stream>>>(x, xb, MM * DD);
  transpose_w_k<<<(DD * 3 * DD + TPB - 1) / TPB, TPB, 0, stream>>>(Wqkv0, w0t, DD, 3 * DD);
  transpose_w_k<<<(DD * 3 * DD + TPB - 1) / TPB, TPB, 0, stream>>>(Wqkv1, w1t, DD, 3 * DD);
  transpose_w_k<<<(2 * DD * DD + TPB - 1) / TPB, TPB, 0, stream>>>(Wg, wgt, 2 * DD, DD);
  transpose_w_k<<<(DD * DD + TPB - 1) / TPB, TPB, 0, stream>>>(Wo, wot, DD, DD);

  const dim3 gblk(256);
  const dim3 ablk(128);
  // pass 1: qkv0 = xb @ Wqkv0 + b ; pred = attend(qkv0)
  gemm_bf16_k<0><<<dim3(MM / 128, 3 * DD / 128), gblk, 0, stream>>>(
      xb, w0t, bqkv0, qkvb, nullptr, nullptr, MM, 3 * DD, DD);
  flash_attn_k<<<dim3(TT / 64, BB * HH), ablk, 0, stream>>>(qkvb, predb);
  // residual = x - pred
  resid_k<<<(MM * DD + TPB - 1) / TPB, TPB, 0, stream>>>(x, predb, residb, MM * DD);
  // pass 2: qkv1 = resid @ Wqkv1 + b ; corr = attend(qkv1)
  gemm_bf16_k<0><<<dim3(MM / 128, 3 * DD / 128), gblk, 0, stream>>>(
      residb, w1t, bqkv1, qkvb, nullptr, nullptr, MM, 3 * DD, DD);
  flash_attn_k<<<dim3(TT / 64, BB * HH), ablk, 0, stream>>>(qkvb, corrb);
  // gate: outb = pred + sigmoid([pred|corr] @ Wg + bg) * corr
  concat_k<<<(MM * 2 * DD + TPB - 1) / TPB, TPB, 0, stream>>>(
      predb, corrb, catb, MM * 2 * DD);
  gemm_bf16_k<1><<<dim3(MM / 128, DD / 128), gblk, 0, stream>>>(
      catb, wgt, bg, outb, predb, corrb, MM, DD, 2 * DD);
  // final: d_out = outb @ Wo + bo  (fp32)
  gemm_bf16_k<2><<<dim3(MM / 128, DD / 128), gblk, 0, stream>>>(
      outb, wot, bo, d_out, nullptr, nullptr, MM, DD, DD);
}
